// Interactor_67439576482318
// MI455X (gfx1250) — matrix operand order repaired and checked by reading, not compiled, not run on hardware
//
#include <hip/hip_runtime.h>
#include <math.h>

typedef _Float16 f16;
typedef __attribute__((ext_vector_type(16))) _Float16 v16h;
typedef __attribute__((ext_vector_type(8)))  _Float16 v8h;
typedef __attribute__((ext_vector_type(8)))  float    v8f;
typedef int v4i __attribute__((vector_size(16)));

#define GG   512
#define NPER 32
#define NN   (GG * NPER)      // 16384
#define DD   128
#define LL   4
#define EE   131072
#define NG   50
#define NGP  64               // padded K for filter GEMM1

#define CUTOFF  10.0f
#define DELTA   (10.0f / 49.0f)
#define GCOEFF  (-0.5f / (DELTA * DELTA))
#define PI_F    3.14159265358979f
#define EPSBN   1e-5f
#define LOG2F_  0.6931471805599453f

#define ACT_NONE 0
#define ACT_RELU 1
#define ACT_SSP  2

#if defined(__has_builtin)
#if __has_builtin(__builtin_amdgcn_global_load_async_to_lds_b128)
#define USE_ASYNC_LDS 1
#endif
#endif

__device__ __forceinline__ float sspf(float x) {
  float sp = (x > 20.0f) ? x : __logf(1.0f + __expf(x));
  return sp - LOG2F_;
}

// copy 16 bytes (8 f16) global -> LDS; async on gfx1250 when available
__device__ __forceinline__ void lds_cp16(f16* dst, const f16* src) {
#ifdef USE_ASYNC_LDS
  f16* s = const_cast<f16*>(src);
  __builtin_amdgcn_global_load_async_to_lds_b128(
      (__attribute__((address_space(1))) v4i*)s,
      (__attribute__((address_space(3))) v4i*)dst, 0, 0);
#else
  *(v8h*)dst = *(const v8h*)src;
#endif
}
__device__ __forceinline__ void lds_cp_wait() {
#ifdef USE_ASYNC_LDS
#if __has_builtin(__builtin_amdgcn_s_wait_asynccnt)
  __builtin_amdgcn_s_wait_asynccnt(0);
#else
  asm volatile("s_wait_asynccnt 0x0" ::: "memory");
#endif
#endif
}

// build a v16h WMMA fragment from two contiguous 16-byte LDS reads
__device__ __forceinline__ v16h frag16(const f16* lo, const f16* hi) {
  v8h a = *(const v8h*)lo;
  v8h b = *(const v8h*)hi;
  return __builtin_shufflevector(a, b, 0, 1, 2, 3, 4, 5, 6, 7,
                                 8, 9, 10, 11, 12, 13, 14, 15);
}

// ---------------------------------------------------------------------------
// weight convert + transpose:  wt[n*K + k] = (f16) w[k*Nc + n]
// ---------------------------------------------------------------------------
__global__ void wcvt_kernel(const float* __restrict__ w, f16* __restrict__ wt,
                            int K, int Nc) {
  int idx = blockIdx.x * blockDim.x + threadIdx.x;
  if (idx >= K * Nc) return;
  int k = idx / Nc, n = idx - k * Nc;
  wt[n * K + k] = (f16)w[idx];
}
// mw1 (NG x D) -> padded transposed f16 [D][NGP]
__global__ void wcvt_mw1_kernel(const float* __restrict__ w, f16* __restrict__ wt) {
  int idx = blockIdx.x * blockDim.x + threadIdx.x;   // DD*NGP
  int c = idx / NGP, r = idx - c * NGP;
  wt[idx] = (f16)(r < NG ? w[r * DD + c] : 0.0f);
}

// ---------------------------------------------------------------------------
// encode: x2d = E1[x0]+E2[x1], x3d = E3[x0]; prev = x
// ---------------------------------------------------------------------------
__global__ void encode_kernel(const int* __restrict__ x,
                              const float* __restrict__ e1,
                              const float* __restrict__ e2,
                              const float* __restrict__ e3,
                              float* x2d, float* x3d, float* p2, float* p3) {
  int n = blockIdx.x, f = threadIdx.x;
  int t0 = x[n * 2 + 0], t1 = x[n * 2 + 1];
  float v2 = e1[t0 * DD + f] + e2[t1 * DD + f];
  float v3 = e3[t0 * DD + f];
  x2d[n * DD + f] = v2; p2[n * DD + f] = v2;
  x3d[n * DD + f] = v3; p3[n * DD + f] = v3;
}

__global__ void zero_kernel(float* p, int n) {
  int i = blockIdx.x * blockDim.x + threadIdx.x;
  if (i < n) p[i] = 0.0f;
}
__global__ void copy_kernel(float* dst, const float* src, int n) {
  int i = blockIdx.x * blockDim.x + threadIdx.x;
  if (i < n) dst[i] = src[i];
}

// ---------------------------------------------------------------------------
// GIN edge scatter-add:  aggr[dst] += x2d[src] + ee1[a0] + ee2[a1]
// ---------------------------------------------------------------------------
__global__ void gin_edge_kernel(const float* __restrict__ x2d,
                                const int* __restrict__ ei,
                                const int* __restrict__ ea,
                                const float* __restrict__ ee1,
                                const float* __restrict__ ee2,
                                float* aggr) {
  int e = blockIdx.x, f = threadIdx.x;
  int src = ei[e], dst = ei[EE + e];
  int a0 = ea[e * 2 + 0], a1 = ea[e * 2 + 1];
  float v = x2d[src * DD + f] + ee1[a0 * DD + f] + ee2[a1 * DD + f];
  atomicAdd(&aggr[dst * DD + f], v);
}

// self-loop term: aggr += x2d + (ee1[4] + ee2[0])
__global__ void gin_self_kernel(float* aggr, const float* __restrict__ x2d,
                                const float* __restrict__ ee1,
                                const float* __restrict__ ee2) {
  int idx = blockIdx.x * blockDim.x + threadIdx.x;
  int f = idx & (DD - 1);
  aggr[idx] += x2d[idx] + ee1[4 * DD + f] + ee2[0 * DD + f];
}

// ---------------------------------------------------------------------------
// Tiled WMMA GEMM:  C[M x Nc] = act(A[M x K] @ B[K x Nc] + bias)
// A: f32 row-major.  Bt: f16, pre-transposed [Nc][K].
// 128x128 block tile, 8 waves (4x2), each wave 32x64 (2x4 WMMA tiles).
// Outputs: Cf (f32) and/or Ch (f16), either may be null.
// Requires M%128==0, Nc%128==0, K%32==0.
// ---------------------------------------------------------------------------
#define BM 128
#define BN 128
#define BK 32

__global__ __launch_bounds__(256) void gemm_wmma_kernel(
    const float* __restrict__ A, const f16* __restrict__ Bt,
    const float* __restrict__ bias, float* __restrict__ Cf,
    f16* __restrict__ Ch, int M, int K, int Nc, int act) {
  __shared__ __align__(32) f16 sA[BM][BK];    // 8 KB
  __shared__ __align__(32) f16 sBt[BN][BK];   // 8 KB

  int tid = threadIdx.x;
  int wid = tid >> 5, lane = tid & 31, half = lane >> 4, l16 = lane & 15;
  int nb = Nc / BN;
  int bm = blockIdx.x / nb, bn = blockIdx.x - bm * nb;
  int m0 = bm * BM, n0 = bn * BN;
  int wm = wid >> 1, wn = wid & 1;   // 4 x 2 wave grid

  v8f acc[2][4];
#pragma unroll
  for (int c = 0; c < 4; ++c) {
    float bv = bias ? bias[n0 + wn * 64 + c * 16 + l16] : 0.0f;
#pragma unroll
    for (int rr = 0; rr < 2; ++rr)
#pragma unroll
      for (int r = 0; r < 8; ++r) acc[rr][c][r] = bv;
  }

  for (int k0 = 0; k0 < K; k0 += BK) {
    // --- stage A tile (f32 -> f16): thread t -> row t/2, 16 contiguous cols
    {
      int r = tid >> 1, cb = (tid & 1) * 16;
      const float* src = A + (size_t)(m0 + r) * K + k0 + cb;
#pragma unroll
      for (int c = 0; c < 16; ++c) sA[r][cb + c] = (f16)src[c];
    }
    // --- stage Bt tile: raw 16B copies (async-to-LDS when available)
    {
#pragma unroll
      for (int s = 0; s < 2; ++s) {
        int idx = tid * 2 + s;            // 512 x 16B segments
        int n = idx >> 2, cb = (idx & 3) * 8;
        lds_cp16(&sBt[n][cb], Bt + (size_t)(n0 + n) * K + k0 + cb);
      }
      lds_cp_wait();
    }
    __syncthreads();

    if (k0 + BK < K && lane == 0) {   // gfx1250 global_prefetch of next A chunk
      __builtin_prefetch(A + (size_t)(m0 + wid * 16) * K + k0 + BK, 0, 3);
    }

    v16h af[2];
#pragma unroll
    for (int rr = 0; rr < 2; ++rr) {
      int row = wm * 32 + rr * 16 + l16;
      af[rr] = frag16(&sA[row][half * 8], &sA[row][16 + half * 8]);
    }
#pragma unroll
    for (int c = 0; c < 4; ++c) {
      int col = wn * 64 + c * 16 + l16;
      v16h bf = frag16(&sBt[col][half * 8], &sBt[col][16 + half * 8]);
#pragma unroll
      for (int rr = 0; rr < 2; ++rr)
        acc[rr][c] = __builtin_amdgcn_wmma_f32_16x16x32_f16(
            false, af[rr], false, bf, (short)0, acc[rr][c], false, false);
    }
    __syncthreads();
  }

#pragma unroll
  for (int rr = 0; rr < 2; ++rr)
#pragma unroll
    for (int c = 0; c < 4; ++c)
#pragma unroll
      for (int r = 0; r < 8; ++r) {
        float v = acc[rr][c][r];
        if (act == ACT_RELU) v = fmaxf(v, 0.0f);
        else if (act == ACT_SSP) v = sspf(v);
        size_t off = (size_t)(m0 + wm * 32 + rr * 16 + r + 8 * half) * Nc +
                     n0 + wn * 64 + c * 16 + l16;
        if (Cf) Cf[off] = v;
        if (Ch) Ch[off] = (f16)v;
      }
}

// ---------------------------------------------------------------------------
// Fused SchNet CFConv: per-graph workgroup recomputes the dense filter in LDS
// (gauss -> WMMA -> ssp -> WMMA) and accumulates agg[i,f] += W*Cm*h1[j,f]
// Weights arrive pre-transposed f16; staging = raw LDS fills (async-capable).
// ---------------------------------------------------------------------------
__global__ __launch_bounds__(256) void cfconv_kernel(
    const float* __restrict__ pos, const f16* __restrict__ h1h,
    const f16* __restrict__ mw1t, const float* __restrict__ mb1,
    const f16* __restrict__ mw2t, const float* __restrict__ mb2,
    float* __restrict__ agg) {
  int g = blockIdx.x;
  int tid = threadIdx.x;
  int wid = tid >> 5, lane = tid & 31, half = lane >> 4, l16 = lane & 15;

  __shared__ float sPos[NPER][3];
  __shared__ float sDist[NPER][NPER];
  __shared__ float sCm[NPER][NPER];
  __shared__ __align__(32) f16 sH1[NPER][DD];   // 8 KB
  __shared__ __align__(32) f16 sW1t[DD][NGP];   // 16 KB
  __shared__ __align__(32) f16 sW2t[DD][DD];    // 32 KB
  __shared__ float sB1[DD], sB2[DD];
  __shared__ __align__(32) f16 sG[64][NGP];     // 8 KB
  __shared__ __align__(32) f16 sT1[64][DD];     // 16 KB

  // raw f16 fills (16B segments): W1t 1024 segs, W2t 2048, H1 512
  {
    f16* w1d = &sW1t[0][0];
    f16* w2d = &sW2t[0][0];
    f16* h1d = &sH1[0][0];
    const f16* h1s = h1h + (size_t)g * NPER * DD;
#pragma unroll
    for (int s = 0; s < 4; ++s)
      lds_cp16(w1d + (tid * 4 + s) * 8, mw1t + (tid * 4 + s) * 8);
#pragma unroll
    for (int s = 0; s < 8; ++s)
      lds_cp16(w2d + (tid * 8 + s) * 8, mw2t + (tid * 8 + s) * 8);
#pragma unroll
    for (int s = 0; s < 2; ++s)
      lds_cp16(h1d + (tid * 2 + s) * 8, h1s + (tid * 2 + s) * 8);
    lds_cp_wait();
  }
  for (int idx = tid; idx < NPER * 3; idx += 256)
    sPos[idx / 3][idx % 3] = pos[g * NPER * 3 + idx];
  for (int idx = tid; idx < DD; idx += 256) { sB1[idx] = mb1[idx]; sB2[idx] = mb2[idx]; }
  __syncthreads();

  for (int idx = tid; idx < NPER * NPER; idx += 256) {
    int ii = idx / NPER, jj = idx % NPER;
    float dx = sPos[ii][0] - sPos[jj][0];
    float dy = sPos[ii][1] - sPos[jj][1];
    float dz = sPos[ii][2] - sPos[jj][2];
    float d = sqrtf(dx * dx + dy * dy + dz * dz + 1e-12f);
    sDist[ii][jj] = d;
    float m = (d < CUTOFF && ii != jj) ? 1.0f : 0.0f;
    float Cc = 0.5f * (__cosf(d * (PI_F / CUTOFF)) + 1.0f);
    sCm[ii][jj] = Cc * m;
  }

  // 16 chunks of 2 source rows i (64 (i,j) pairs per chunk)
  for (int ci = 0; ci < 16; ++ci) {
    int i0 = ci * 2;
    __syncthreads();
    for (int idx = tid; idx < 64 * NGP; idx += 256) {
      int p = idx / NGP, c = idx % NGP;
      int ii = i0 + (p >> 5), jj = p & 31;
      float dc = sDist[ii][jj] - (float)c * DELTA;
      sG[p][c] = (f16)((c < NG) ? __expf(GCOEFF * dc * dc) : 0.0f);
    }
    __syncthreads();
    // GEMM1: t1 = ssp(gauss[64x64] @ mw1[64x128] + b1)   (4x8 = 32 tiles)
    for (int t = wid; t < 32; t += 8) {
      int tm = t >> 3, tn = t & 7;
      v8f acc;
      float bv = sB1[tn * 16 + l16];
#pragma unroll
      for (int r = 0; r < 8; ++r) acc[r] = bv;
#pragma unroll
      for (int k0 = 0; k0 < NGP; k0 += 32) {
        int row = tm * 16 + l16, col = tn * 16 + l16;
        v16h a = frag16(&sG[row][k0 + half * 8], &sG[row][k0 + 16 + half * 8]);
        v16h b = frag16(&sW1t[col][k0 + half * 8], &sW1t[col][k0 + 16 + half * 8]);
        acc = __builtin_amdgcn_wmma_f32_16x16x32_f16(false, a, false, b,
                                                     (short)0, acc, false, false);
      }
#pragma unroll
      for (int r = 0; r < 8; ++r)
        sT1[tm * 16 + r + 8 * half][tn * 16 + l16] = (f16)sspf(acc[r]);
    }
    __syncthreads();
    // GEMM2 + cutoff mask + accumulate into agg (global f32 atomics)
    for (int t = wid; t < 32; t += 8) {
      int tm = t >> 3, tn = t & 7;
      v8f acc;
      float bv = sB2[tn * 16 + l16];
#pragma unroll
      for (int r = 0; r < 8; ++r) acc[r] = bv;
#pragma unroll
      for (int k0 = 0; k0 < DD; k0 += 32) {
        int row = tm * 16 + l16, col = tn * 16 + l16;
        v16h a = frag16(&sT1[row][k0 + half * 8], &sT1[row][k0 + 16 + half * 8]);
        v16h b = frag16(&sW2t[col][k0 + half * 8], &sW2t[col][k0 + 16 + half * 8]);
        acc = __builtin_amdgcn_wmma_f32_16x16x32_f16(false, a, false, b,
                                                     (short)0, acc, false, false);
      }
#pragma unroll
      for (int r = 0; r < 8; ++r) {
        int p = tm * 16 + r + 8 * half;
        int ii = i0 + (p >> 5), jj = p & 31;
        int f = tn * 16 + l16;
        float w = acc[r] * sCm[ii][jj];
        atomicAdd(&agg[(g * NPER + ii) * DD + f], w * (float)sH1[jj][f]);
      }
    }
  }
}

// ---------------------------------------------------------------------------
// BatchNorm (training-mode batch stats over rows): one block per feature
// ---------------------------------------------------------------------------
__global__ __launch_bounds__(256) void bn_stats_kernel(
    const float* __restrict__ X, int M, int C, float* mean, float* rvar) {
  int f = blockIdx.x, tid = threadIdx.x;
  float s = 0.0f, s2 = 0.0f;
  for (int n = tid; n < M; n += 256) {
    float v = X[n * C + f];
    s += v; s2 += v * v;
  }
  __shared__ float rs[256], rq[256];
  rs[tid] = s; rq[tid] = s2;
  __syncthreads();
  for (int st = 128; st > 0; st >>= 1) {
    if (tid < st) { rs[tid] += rs[tid + st]; rq[tid] += rq[tid + st]; }
    __syncthreads();
  }
  if (tid == 0) {
    float mu = rs[0] / (float)M;
    float var = rq[0] / (float)M - mu * mu;
    mean[f] = mu;
    rvar[f] = rsqrtf(var + EPSBN);
  }
}

// X = relu((H-mu)*rvar*g + b) [+ prev]   (cmask = C-1, C power of two)
__global__ void bn_apply_kernel(const float* __restrict__ H,
                                const float* __restrict__ mean,
                                const float* __restrict__ rvar,
                                const float* __restrict__ gam,
                                const float* __restrict__ bet,
                                const float* __restrict__ prev,
                                float* __restrict__ X, int cmask) {
  int idx = blockIdx.x * blockDim.x + threadIdx.x;
  int f = idx & cmask;
  float v = (H[idx] - mean[f]) * rvar[f] * gam[f] + bet[f];
  v = fmaxf(v, 0.0f);
  if (prev) v += prev[idx];
  X[idx] = v;
}

// ---------------------------------------------------------------------------
// virtual-node gather / scatter / final output
// ---------------------------------------------------------------------------
__global__ void vgather_kernel(const float* __restrict__ x2d,
                               const float* __restrict__ x3d,
                               float* __restrict__ inter) {
  int g = blockIdx.x, c = threadIdx.x;
  int v = (g * NPER + NPER - 1) * DD;
  inter[g * 256 + c] = (c < DD) ? x2d[v + c] : x3d[v + c - DD];
}
__global__ void vscatter_kernel(const float* __restrict__ z,
                                float* x2d, float* x3d) {
  int g = blockIdx.x, c = threadIdx.x;
  int v = (g * NPER + NPER - 1) * DD;
  if (c < DD) x2d[v + c] = z[g * 256 + c];
  else        x3d[v + c - DD] = z[g * 256 + c];
}
__global__ void out_kernel(const float* __restrict__ x2d,
                           const float* __restrict__ x3d,
                           float* __restrict__ out) {
  int g = blockIdx.x, c = threadIdx.x;
  int v = (g * NPER + NPER - 1) * DD;
  out[g * 256 + c] = (c < DD) ? x2d[v + c] : x3d[v + c - DD];
}

// ---------------------------------------------------------------------------
extern "C" void kernel_launch(void* const* d_in, const int* in_sizes, int n_in,
                              void* d_out, int out_size, void* d_ws, size_t ws_size,
                              hipStream_t stream) {
  const int*   x   = (const int*)d_in[0];
  const int*   ei  = (const int*)d_in[1];
  const int*   ea  = (const int*)d_in[2];
  const float* pos = (const float*)d_in[3];
  // d_in[4] = batch (structural, unused)
  const float* a2d_e1 = (const float*)d_in[5];
  const float* a2d_e2 = (const float*)d_in[6];
  const float* a3d    = (const float*)d_in[7];
  const float* gee1   = (const float*)d_in[8];   // [L,5,D]
  const float* gee2   = (const float*)d_in[9];   // [L,3,D]
  const float* gw1    = (const float*)d_in[10];  // [L,D,2D]
  const float* gb1    = (const float*)d_in[11];  // [L,2D]
  const float* gw2    = (const float*)d_in[12];  // [L,2D,D]
  const float* gb2    = (const float*)d_in[13];  // [L,D]
  const float* smw1   = (const float*)d_in[14];
  const float* smb1   = (const float*)d_in[15];
  const float* smw2   = (const float*)d_in[16];
  const float* smb2   = (const float*)d_in[17];
  const float* slin1  = (const float*)d_in[18];
  const float* slin2  = (const float*)d_in[19];
  const float* slin2b = (const float*)d_in[20];
  const float* sactw  = (const float*)d_in[21];
  const float* sactb  = (const float*)d_in[22];
  const float* ng     = (const float*)d_in[23];
  const float* nb     = (const float*)d_in[24];
  const float* iw1    = (const float*)d_in[25];
  const float* ib1    = (const float*)d_in[26];
  const float* ig     = (const float*)d_in[27];
  const float* ibt    = (const float*)d_in[28];
  const float* iw2    = (const float*)d_in[29];
  const float* ib2    = (const float*)d_in[30];

  const int NT = NN * DD;                 // 2M floats
  float* ws    = (float*)d_ws;
  float* x2d   = ws + 0 * (size_t)NT;
  float* x3d   = ws + 1 * (size_t)NT;
  float* p2    = ws + 2 * (size_t)NT;
  float* p3    = ws + 3 * (size_t)NT;
  float* aggr  = ws + 4 * (size_t)NT;
  float* agg3  = ws + 5 * (size_t)NT;
  float* htmp  = ws + 6 * (size_t)NT;
  float* hbig  = ws + 7 * (size_t)NT;     // N x 256
  float* meanb = ws + 9 * (size_t)NT;
  float* rvarb = meanb + 256;
  float* inter = rvarb + 256;             // G x 256
  float* v1    = inter + GG * 256;
  float* v2    = v1 + GG * 256;
  // f16 area (16B aligned: all offsets are multiples of 4 floats)
  f16* fb      = (f16*)(v2 + GG * 256);
  f16* gw1t    = fb;                        // 4 * 256*128
  f16* gw2t    = gw1t + 4 * 256 * 128;      // 4 * 128*256
  f16* slin1t  = gw2t + 4 * 128 * 256;      // 128*128
  f16* slin2t  = slin1t + DD * DD;
  f16* sactwt  = slin2t + DD * DD;
  f16* iw1t    = sactwt + DD * DD;          // 256*256
  f16* iw2t    = iw1t + 256 * 256;
  f16* mw2t    = iw2t + 256 * 256;          // 128*128
  f16* mw1t    = mw2t + DD * DD;            // DD*NGP
  f16* h1h     = mw1t + DD * NGP;           // N*D f16

  // ---- one-time weight convert/transpose (per launch) ----
  for (int i = 0; i < LL; ++i) {
    wcvt_kernel<<<(DD * 256 + 255) / 256, 256, 0, stream>>>(
        gw1 + (size_t)i * DD * 256, gw1t + (size_t)i * 256 * DD, DD, 256);
    wcvt_kernel<<<(256 * DD + 255) / 256, 256, 0, stream>>>(
        gw2 + (size_t)i * 256 * DD, gw2t + (size_t)i * DD * 256, 256, DD);
  }
  wcvt_kernel<<<(DD * DD + 255) / 256, 256, 0, stream>>>(slin1, slin1t, DD, DD);
  wcvt_kernel<<<(DD * DD + 255) / 256, 256, 0, stream>>>(slin2, slin2t, DD, DD);
  wcvt_kernel<<<(DD * DD + 255) / 256, 256, 0, stream>>>(sactw, sactwt, DD, DD);
  wcvt_kernel<<<(256 * 256 + 255) / 256, 256, 0, stream>>>(iw1, iw1t, 256, 256);
  wcvt_kernel<<<(256 * 256 + 255) / 256, 256, 0, stream>>>(iw2, iw2t, 256, 256);
  wcvt_kernel<<<(DD * DD + 255) / 256, 256, 0, stream>>>(smw2, mw2t, DD, DD);
  wcvt_mw1_kernel<<<(DD * NGP + 255) / 256, 256, 0, stream>>>(smw1, mw1t);

  encode_kernel<<<NN, DD, 0, stream>>>(x, a2d_e1, a2d_e2, a3d, x2d, x3d, p2, p3);

  const int ewB = (NT + 255) / 256;
  for (int i = 0; i < LL; ++i) {
    const float* ee1 = gee1 + (size_t)i * 5 * DD;
    const float* ee2 = gee2 + (size_t)i * 3 * DD;
    const f16*   w1t = gw1t + (size_t)i * 256 * DD;
    const float* b1  = gb1  + (size_t)i * 2 * DD;
    const f16*   w2t = gw2t + (size_t)i * DD * 256;
    const float* b2  = gb2  + (size_t)i * DD;

    // --- 2D GIN ---
    zero_kernel<<<ewB, 256, 0, stream>>>(aggr, NT);
    gin_edge_kernel<<<EE, DD, 0, stream>>>(x2d, ei, ea, ee1, ee2, aggr);
    gin_self_kernel<<<ewB, 256, 0, stream>>>(aggr, x2d, ee1, ee2);
    gemm_wmma_kernel<<<(NN / BM) * (256 / BN), 256, 0, stream>>>(
        aggr, w1t, b1, hbig, nullptr, NN, DD, 2 * DD, ACT_RELU);
    gemm_wmma_kernel<<<(NN / BM) * (DD / BN), 256, 0, stream>>>(
        hbig, w2t, b2, htmp, nullptr, NN, 2 * DD, DD, ACT_NONE);
    bn_stats_kernel<<<DD, 256, 0, stream>>>(htmp, NN, DD, meanb, rvarb);
    bn_apply_kernel<<<ewB, 256, 0, stream>>>(htmp, meanb, rvarb, ng, nb, p2, x2d, DD - 1);

    // --- 3D SchNet interaction ---
    gemm_wmma_kernel<<<(NN / BM) * (DD / BN), 256, 0, stream>>>(
        x3d, slin1t, nullptr, nullptr, h1h, NN, DD, DD, ACT_NONE);
    zero_kernel<<<ewB, 256, 0, stream>>>(agg3, NT);
    cfconv_kernel<<<GG, 256, 0, stream>>>(pos, h1h, mw1t, smb1, mw2t, smb2, agg3);
    gemm_wmma_kernel<<<(NN / BM) * (DD / BN), 256, 0, stream>>>(
        agg3, slin2t, slin2b, hbig, nullptr, NN, DD, DD, ACT_SSP);
    gemm_wmma_kernel<<<(NN / BM) * (DD / BN), 256, 0, stream>>>(
        hbig, sactwt, sactb, htmp, nullptr, NN, DD, DD, ACT_NONE);
    bn_stats_kernel<<<DD, 256, 0, stream>>>(htmp, NN, DD, meanb, rvarb);
    bn_apply_kernel<<<ewB, 256, 0, stream>>>(htmp, meanb, rvarb, ng, nb, p3, x3d, DD - 1);

    // --- virtual-node interactor ---
    vgather_kernel<<<GG, 256, 0, stream>>>(x2d, x3d, inter);
    gemm_wmma_kernel<<<(GG / BM) * (256 / BN), 256, 0, stream>>>(
        inter, iw1t, ib1, v1, nullptr, GG, 256, 256, ACT_NONE);
    bn_stats_kernel<<<256, 256, 0, stream>>>(v1, GG, 256, meanb, rvarb);
    {
      int vwB = (GG * 256 + 255) / 256;
      bn_apply_kernel<<<vwB, 256, 0, stream>>>(v1, meanb, rvarb, ig, ibt, nullptr, v2, 255);
    }
    gemm_wmma_kernel<<<(GG / BM) * (256 / BN), 256, 0, stream>>>(
        v2, iw2t, ib2, v1, nullptr, GG, 256, 256, ACT_NONE);
    vscatter_kernel<<<GG, 256, 0, stream>>>(v1, x2d, x3d);
    copy_kernel<<<ewB, 256, 0, stream>>>(p2, x2d, NT);
    copy_kernel<<<ewB, 256, 0, stream>>>(p3, x3d, NT);
  }

  out_kernel<<<GG, 256, 0, stream>>>(x2d, x3d, (float*)d_out);
}
